// DifferentialAttention_11819749999122
// MI455X (gfx1250) — matrix operand order repaired and checked by reading, not compiled, not run on hardware
//
#include <hip/hip_runtime.h>
#include <math.h>

typedef __bf16 bf16_t;
typedef __attribute__((ext_vector_type(16))) __bf16 v16bf;
typedef __attribute__((ext_vector_type(8)))  float  v8f;

#define LAMBDA_INIT 0.4707130183435842f

__device__ __forceinline__ v8f wmma_bf16(v16bf a, v16bf b, v8f c) {
  return __builtin_amdgcn_wmma_f32_16x16x32_bf16(false, a, false, b, (short)0, c, false, false);
}

// A-fragment (16x32 bf16, MxK) from row-major [m][k] buffer.
// ISA 7.12.2: lanes 0-15 -> M=0..15 with K in {0..7,16..23}; lanes 16-31 same
// rows, K in {8..15,24..31}. Element j<8 -> K=8h+j ; j>=8 -> K=16+8h+(j-8).
__device__ __forceinline__ v16bf frag_A(const bf16_t* base, int stride,
                                        int row0, int koff) {
  int l = threadIdx.x & 31;
  const bf16_t* p = base + (size_t)(row0 + (l & 15)) * stride + koff + 8 * (l >> 4);
  v16bf f;
#pragma unroll
  for (int j = 0; j < 8; ++j) { f[j] = p[j]; f[j + 8] = p[j + 16]; }
  return f;
}

// B-fragment (32x16 bf16, KxN) from a row-major [n][k] buffer (i.e. B^T).
// B layout matching frag_A: element j<8 -> K=8h+j ; j>=8 -> K=16+8h+(j-8).
__device__ __forceinline__ v16bf frag_B(const bf16_t* base, int stride,
                                        int n0, int koff) {
  int l = threadIdx.x & 31;
  const bf16_t* p = base + (size_t)(n0 + (l & 15)) * stride + koff + 8 * (l >> 4);
  v16bf f;
#pragma unroll
  for (int j = 0; j < 8; ++j) { f[j] = p[j]; f[j + 8] = p[j + 16]; }
  return f;
}

// ---------------------------------------------------------------------------
// lambda_full = exp(sum lq1*lk1) - exp(sum lq2*lk2) + LAMBDA_INIT
// ---------------------------------------------------------------------------
__global__ void lambda_kernel(const float* __restrict__ q1, const float* __restrict__ k1,
                              const float* __restrict__ q2, const float* __restrict__ k2,
                              float* __restrict__ lam) {
  __shared__ float s1[64], s2[64];
  int t = threadIdx.x;
  s1[t] = q1[t] * k1[t];
  s2[t] = q2[t] * k2[t];
  __syncthreads();
  if (t == 0) {
    float a = 0.f, b = 0.f;
    for (int i = 0; i < 64; ++i) { a += s1[i]; b += s2[i]; }
    *lam = expf(a) - expf(b) + LAMBDA_INIT;
  }
}

// ---------------------------------------------------------------------------
// C[M,N] (fp32) = A[M,K] (fp32, row-major) x Bw[N,K]^T (fp32, row-major)
// via bf16 WMMA, fp32 accumulate. Tile 128x128, K-chunk 64, 8 waves (2x4).
// ---------------------------------------------------------------------------
#define G_STR 72  // 64 + 8 pad (bf16 elems)

__global__ __launch_bounds__(256) void gemm_nt_bf16(const float* __restrict__ A,
                                                    const float* __restrict__ Bw,
                                                    float* __restrict__ C,
                                                    int M, int N, int K) {
  __shared__ bf16_t sA[128 * G_STR];
  __shared__ bf16_t sB[128 * G_STR];
  int tid = threadIdx.x;
  int wid = tid >> 5;
  int wm = wid >> 2;   // 0..1 -> 64 rows each
  int wn = wid & 3;    // 0..3 -> 32 cols each
  int tile_m = blockIdx.y * 128;
  int tile_n = blockIdx.x * 128;

  v8f acc[4][2];
#pragma unroll
  for (int i = 0; i < 4; ++i)
#pragma unroll
    for (int j = 0; j < 2; ++j) acc[i][j] = {};

  for (int kk = 0; kk < K; kk += 64) {
    // ---- load phase: batch all 16 global b128 loads (max MLP, no per-load
    //      full waits), then convert/store phase into LDS.
    float4 va[8], vb[8];
#pragma unroll
    for (int it = 0; it < 8; ++it) {
      int slot = tid + it * 256;            // 0..2047 float4 slots
      int row = slot >> 4;                  // 0..127
      int kof = (slot & 15) * 4;            // 0..60
      va[it] = *(const float4*)(A + (size_t)(tile_m + row) * K + kk + kof);
      vb[it] = *(const float4*)(Bw + (size_t)(tile_n + row) * K + kk + kof);
    }
    if (kk + 64 < K) {  // emits global_prefetch_b8
      __builtin_prefetch(A + (size_t)(tile_m + (tid >> 1)) * K + kk + 64, 0, 0);
      __builtin_prefetch(Bw + (size_t)(tile_n + (tid >> 1)) * K + kk + 64, 0, 0);
    }
#pragma unroll
    for (int it = 0; it < 8; ++it) {
      int slot = tid + it * 256;
      int row = slot >> 4;
      int kof = (slot & 15) * 4;
      bf16_t* da = sA + row * G_STR + kof;
      da[0] = (bf16_t)va[it].x; da[1] = (bf16_t)va[it].y;
      da[2] = (bf16_t)va[it].z; da[3] = (bf16_t)va[it].w;
      bf16_t* db = sB + row * G_STR + kof;
      db[0] = (bf16_t)vb[it].x; db[1] = (bf16_t)vb[it].y;
      db[2] = (bf16_t)vb[it].z; db[3] = (bf16_t)vb[it].w;
    }
    __syncthreads();

#pragma unroll
    for (int kc = 0; kc < 2; ++kc) {
      v16bf aF[4], bF[2];
#pragma unroll
      for (int ms = 0; ms < 4; ++ms)
        aF[ms] = frag_A(sA, G_STR, wm * 64 + ms * 16, kc * 32);
#pragma unroll
      for (int ns = 0; ns < 2; ++ns)
        bF[ns] = frag_B(sB, G_STR, wn * 32 + ns * 16, kc * 32);
#pragma unroll
      for (int ms = 0; ms < 4; ++ms)
#pragma unroll
        for (int ns = 0; ns < 2; ++ns)
          acc[ms][ns] = wmma_bf16(aF[ms], bF[ns], acc[ms][ns]);
    }
    __syncthreads();
  }

  // C-fragment layout: VGPR r -> M = r + 8*(lane>>4); N = lane&15
  int l = tid & 31;
  int h = l >> 4;
  int cn = l & 15;
#pragma unroll
  for (int ms = 0; ms < 4; ++ms)
#pragma unroll
    for (int ns = 0; ns < 2; ++ns) {
      float* cp = C + (size_t)(tile_m + wm * 64 + ms * 16 + 8 * h) * N +
                  tile_n + wn * 32 + ns * 16 + cn;
#pragma unroll
      for (int r = 0; r < 8; ++r) cp[(size_t)r * N] = acc[ms][ns][r];
    }
}

// ---------------------------------------------------------------------------
// RoPE + reorder to per-head layouts, fp32 -> bf16
// ---------------------------------------------------------------------------
__global__ __launch_bounds__(256) void prep_q(const float* __restrict__ qb,
                                              bf16_t* __restrict__ qh, int S) {
  int lin = blockIdx.x * 256 + threadIdx.x;   // over S*64
  int s = lin >> 6, d = lin & 63;
  int head = blockIdx.y, b = blockIdx.z;
  size_t row = (size_t)(b * S + s);
  float x = qb[row * 2048 + head * 64 + d];
  float o = qb[row * 2048 + head * 64 + ((d < 32) ? d + 32 : d - 32)];
  float rot = (d < 32) ? -o : o;
  float inv = __powf(10000.0f, -(float)d * (1.0f / 64.0f));
  float sn, cs;
  __sincosf((float)s * inv, &sn, &cs);
  float val = (x * cs + rot * sn) * 0.125f;   // * Dh^-0.5
  qh[(((size_t)b * 32 + head) * S + s) * 64 + d] = (bf16_t)val;
}

__global__ __launch_bounds__(256) void prep_k(const float* __restrict__ kb,
                                              bf16_t* __restrict__ kh, int S) {
  int lin = blockIdx.x * 256 + threadIdx.x;
  int s = lin >> 6, d = lin & 63;
  int head = blockIdx.y, b = blockIdx.z;
  size_t row = (size_t)(b * S + s);
  float x = kb[row * 1024 + head * 64 + d];
  float o = kb[row * 1024 + head * 64 + ((d < 32) ? d + 32 : d - 32)];
  float rot = (d < 32) ? -o : o;
  float inv = __powf(10000.0f, -(float)d * (1.0f / 64.0f));
  float sn, cs;
  __sincosf((float)s * inv, &sn, &cs);
  kh[(((size_t)b * 16 + head) * S + s) * 64 + d] = (bf16_t)(x * cs + rot * sn);
}

// V transposed per head: vt[(b*8+h)*128 + d][s] (bf16) from v_buf fp32.
// LDS-tiled 32x32 transpose, coalesced on both sides.
__global__ __launch_bounds__(256) void prep_v_t(const float* __restrict__ vb,
                                                bf16_t* __restrict__ vt, int S) {
  __shared__ float t[32][33];
  int s0 = blockIdx.x * 32;
  int d0 = blockIdx.y * 32;
  int bh = blockIdx.z;  // b*8 + head
  int b = bh >> 3, head = bh & 7;
  int tx = threadIdx.x & 31, ty = threadIdx.x >> 5;  // 32 x 8
#pragma unroll
  for (int i = 0; i < 32; i += 8)
    t[ty + i][tx] = vb[(size_t)(b * S + s0 + ty + i) * 1024 + head * 128 + d0 + tx];
  __syncthreads();
#pragma unroll
  for (int i = 0; i < 32; i += 8)
    vt[((size_t)bh * 128 + d0 + ty + i) * S + s0 + tx] = (bf16_t)t[tx][ty + i];
}

// ---------------------------------------------------------------------------
// Flash attention: one block per (b, score-head, 64-row q tile), 4 waves.
// O[b,i,s,:] = softmax(Q_i K_{i/2}^T + causal) V_{i/4}, fp32 out.
// ---------------------------------------------------------------------------
#define K_STR 72   // 64 + 8
#define V_STR 72   // 64 + 8 (V^T tile: 128 rows of 64 keys)

__global__ __launch_bounds__(128) void diff_flash_attn(const bf16_t* __restrict__ qh,
                                                       const bf16_t* __restrict__ kh,
                                                       const bf16_t* __restrict__ vt,
                                                       float* __restrict__ O, int S) {
  __shared__ bf16_t sK[64 * K_STR];    // [key][d]   -> B^T for Q.K^T
  __shared__ bf16_t sVt[128 * V_STR];  // [n][key]   -> B^T for P.V
  __shared__ bf16_t sP[4 * 16 * K_STR];

  int tid = threadIdx.x, wid = tid >> 5, l = tid & 31, h = l >> 4, ln = l & 15;
  int qt = blockIdx.x, head = blockIdx.y, b = blockIdx.z;
  const bf16_t* Qp = qh + (((size_t)b * 32 + head) * S + qt * 64) * 64;
  const bf16_t* Kp = kh + (((size_t)b * 16 + (head >> 1)) * S) * 64;
  const bf16_t* Vt = vt + ((size_t)(b * 8 + (head >> 2)) * 128) * S;
  float* Op = O + (((size_t)b * 32 + head) * S + qt * 64) * 128;

  // per-wave 16 Q rows, resident A-fragments (K=64 -> 2 chunks)
  v16bf qf[2];
  qf[0] = frag_A(Qp + (size_t)(wid * 16) * 64, 64, 0, 0);
  qf[1] = frag_A(Qp + (size_t)(wid * 16) * 64, 64, 0, 32);

  float m_i[8], l_i[8];
  v8f accO[8];
#pragma unroll
  for (int r = 0; r < 8; ++r) { m_i[r] = -1e30f; l_i[r] = 0.f; }
#pragma unroll
  for (int nv = 0; nv < 8; ++nv) accO[nv] = {};

  for (int kt = 0; kt <= qt; ++kt) {
    // ---- load phase: batch 4 (K tile) + 8 (V^T tile) global b128 loads
    const uint4* gk = (const uint4*)(Kp + (size_t)kt * 64 * 64);
    uint4 kreg[4], vreg[8];
#pragma unroll
    for (int it = 0; it < 4; ++it) kreg[it] = gk[tid + it * 128];
#pragma unroll
    for (int it = 0; it < 8; ++it) {
      int slot = tid + it * 128;           // 0..1023
      int row = slot >> 3, c8 = (slot & 7) * 8;
      vreg[it] = *(const uint4*)(Vt + (size_t)row * S + kt * 64 + c8);
    }
    // ---- store phase into LDS
#pragma unroll
    for (int it = 0; it < 4; ++it) {
      int slot = tid + it * 128;           // 0..511
      int row = slot >> 3, c8 = (slot & 7) * 8;
      *(uint4*)(sK + row * K_STR + c8) = kreg[it];
    }
#pragma unroll
    for (int it = 0; it < 8; ++it) {
      int slot = tid + it * 128;
      int row = slot >> 3, c8 = (slot & 7) * 8;
      *(uint4*)(sVt + row * V_STR + c8) = vreg[it];
    }
    __syncthreads();

    // scores S = Q K^T  (16 rows x 64 cols per wave)
    v8f sc[4];
#pragma unroll
    for (int ns = 0; ns < 4; ++ns) {
      sc[ns] = {};
#pragma unroll
      for (int kc = 0; kc < 2; ++kc) {
        v16bf kf = frag_B(sK, K_STR, ns * 16, kc * 32);
        sc[ns] = wmma_bf16(qf[kc], kf, sc[ns]);
      }
    }

    if (kt == qt) {  // causal mask on diagonal tile
      int qrow = qt * 64 + wid * 16 + 8 * h;
      int col = kt * 64 + ln;
#pragma unroll
      for (int ns = 0; ns < 4; ++ns)
#pragma unroll
        for (int r = 0; r < 8; ++r)
          if (col + ns * 16 > qrow + r) sc[ns][r] = -1e30f;
    }

    // online softmax per row (cols spread over 16 lanes of each half-wave)
    float alpha[8];
#pragma unroll
    for (int r = 0; r < 8; ++r) {
      float mx = sc[0][r];
#pragma unroll
      for (int ns = 1; ns < 4; ++ns) mx = fmaxf(mx, sc[ns][r]);
#pragma unroll
      for (int off = 1; off < 16; off <<= 1) mx = fmaxf(mx, __shfl_xor(mx, off, 32));
      float mn = fmaxf(m_i[r], mx);
      alpha[r] = __expf(m_i[r] - mn);
      m_i[r] = mn;
      float rs = 0.f;
#pragma unroll
      for (int ns = 0; ns < 4; ++ns) {
        float p = __expf(sc[ns][r] - mn);
        sc[ns][r] = p;
        rs += p;
      }
#pragma unroll
      for (int off = 1; off < 16; off <<= 1) rs += __shfl_xor(rs, off, 32);
      l_i[r] = l_i[r] * alpha[r] + rs;
    }
#pragma unroll
    for (int nv = 0; nv < 8; ++nv)
#pragma unroll
      for (int r = 0; r < 8; ++r) accO[nv][r] *= alpha[r];

    // C-layout P -> wave-private LDS -> A-layout fragments
    bf16_t* pw = sP + wid * 16 * K_STR;
#pragma unroll
    for (int ns = 0; ns < 4; ++ns)
#pragma unroll
      for (int r = 0; r < 8; ++r)
        pw[(r + 8 * h) * K_STR + ns * 16 + ln] = (bf16_t)sc[ns][r];
    asm volatile("s_wait_dscnt 0" ::: "memory");

    // O += P V
#pragma unroll
    for (int kc = 0; kc < 2; ++kc) {
      v16bf pf = frag_A(pw, K_STR, 0, kc * 32);
#pragma unroll
      for (int nv = 0; nv < 8; ++nv) {
        v16bf vf = frag_B(sVt, V_STR, nv * 16, kc * 32);
        accO[nv] = wmma_bf16(pf, vf, accO[nv]);
      }
    }
    __syncthreads();
  }

  float inv[8];
#pragma unroll
  for (int r = 0; r < 8; ++r) inv[r] = 1.0f / l_i[r];
#pragma unroll
  for (int nv = 0; nv < 8; ++nv)
#pragma unroll
    for (int r = 0; r < 8; ++r)
      Op[(size_t)(wid * 16 + 8 * h + r) * 128 + nv * 16 + ln] = accO[nv][r] * inv[r];
}

// ---------------------------------------------------------------------------
// diff + RMSNorm: out[b,s,j*128+d] = norm(O[b,2j,s,d] - lam*O[b,2j+1,s,d])
// ---------------------------------------------------------------------------
__global__ __launch_bounds__(128) void diff_rmsnorm(const float* __restrict__ O,
                                                    const float* __restrict__ subw,
                                                    const float* __restrict__ lam,
                                                    float* __restrict__ out, int S) {
  int d = threadIdx.x;  // 0..127
  int s = blockIdx.x, j = blockIdx.y, b = blockIdx.z;
  float lf = *lam;
  const float* o0 = O + (((size_t)b * 32 + 2 * j) * S + s) * 128;
  const float* o1 = O + (((size_t)b * 32 + 2 * j + 1) * S + s) * 128;
  float x = o0[d] - lf * o1[d];
  float ss = x * x;
#pragma unroll
  for (int off = 1; off < 32; off <<= 1) ss += __shfl_xor(ss, off, 32);
  __shared__ float red[4];
  int wid = d >> 5;
  if ((d & 31) == 0) red[wid] = ss;
  __syncthreads();
  float tot = red[0] + red[1] + red[2] + red[3];
  float y = x * rsqrtf(tot * (1.0f / 128.0f) + 1e-5f) * subw[d] * (1.0f - LAMBDA_INIT);
  out[(size_t)(b * S + s) * 2048 + j * 128 + d] = y;
}

// ---------------------------------------------------------------------------
extern "C" void kernel_launch(void* const* d_in, const int* in_sizes, int n_in,
                              void* d_out, int out_size, void* d_ws, size_t ws_size,
                              hipStream_t stream) {
  const float* hidden = (const float*)d_in[0];
  const float* Wq = (const float*)d_in[1];
  const float* Wk = (const float*)d_in[2];
  const float* Wv = (const float*)d_in[3];
  const float* Wo = (const float*)d_in[4];
  const float* lq1 = (const float*)d_in[5];
  const float* lk1 = (const float*)d_in[6];
  const float* lq2 = (const float*)d_in[7];
  const float* lk2 = (const float*)d_in[8];
  const float* subw = (const float*)d_in[9];
  float* out = (float*)d_out;

  const int Bb = 2, S = 2048, M = Bb * S;  // M = 4096
  char* ws = (char*)d_ws;
  // Region A (67.1MB): q_buf/k_buf/v_buf fp32, later aliased by O
  // Region B (33.6MB): qh/kh/vt bf16, later aliased by attn_flat
  const size_t offA = 0;
  const size_t offB = 67108864;
  const size_t offL = offB + 33554432;

  float* q_buf = (float*)(ws + offA);              // 4096x2048 fp32 = 32MB
  float* k_buf = (float*)(ws + offA + 33554432);   // 4096x1024 fp32 = 16MB
  float* v_buf = (float*)(ws + offA + 50331648);   // 4096x1024 fp32 = 16MB
  float* Obuf  = (float*)(ws + offA);              // [2,32,2048,128] fp32 = 64MB (aliases qkv)
  bf16_t* qh  = (bf16_t*)(ws + offB);              // [2,32,2048,64] bf16 = 16MB
  bf16_t* khp = (bf16_t*)(ws + offB + 16777216);   // [2,16,2048,64] bf16 = 8MB
  bf16_t* vtp = (bf16_t*)(ws + offB + 25165824);   // [2,8,128,2048] bf16 = 8MB (V^T)
  float* attn_flat = (float*)(ws + offB);          // 4096x2048 fp32 = 32MB (aliases qh/kh/vt)
  float* lam = (float*)(ws + offL);

  lambda_kernel<<<1, 64, 0, stream>>>(lq1, lk1, lq2, lk2, lam);

  gemm_nt_bf16<<<dim3(2048 / 128, M / 128), 256, 0, stream>>>(hidden, Wq, q_buf, M, 2048, 2048);
  gemm_nt_bf16<<<dim3(1024 / 128, M / 128), 256, 0, stream>>>(hidden, Wk, k_buf, M, 1024, 2048);
  gemm_nt_bf16<<<dim3(1024 / 128, M / 128), 256, 0, stream>>>(hidden, Wv, v_buf, M, 1024, 2048);

  prep_q<<<dim3(S * 64 / 256, 32, Bb), 256, 0, stream>>>(q_buf, qh, S);
  prep_k<<<dim3(S * 64 / 256, 16, Bb), 256, 0, stream>>>(k_buf, khp, S);
  prep_v_t<<<dim3(S / 32, 4, Bb * 8), 256, 0, stream>>>(v_buf, vtp, S);

  diff_flash_attn<<<dim3(S / 64, 32, Bb), 128, 0, stream>>>(qh, khp, vtp, Obuf, S);

  diff_rmsnorm<<<dim3(S, 16, Bb), 128, 0, stream>>>(Obuf, subw, lam, attn_flat, S);

  gemm_nt_bf16<<<dim3(2048 / 128, M / 128), 256, 0, stream>>>(attn_flat, Wo, out, M, 2048, 2048);
}
